// GPT_28845000360016
// MI455X (gfx1250) — matrix operand order repaired and checked by reading, not compiled, not run on hardware
//
#include <hip/hip_runtime.h>
#include <hip/hip_bf16.h>
#include <math.h>

// ---------------------------------------------------------------------------
// GPT-2 forward for MI455X (gfx1250): wave32, V_WMMA_F32_16X16X32_BF16,
// async global->LDS staging (ASYNCcnt) with double-buffered tiles.
// ---------------------------------------------------------------------------

typedef __bf16 bf16;
typedef __attribute__((ext_vector_type(16))) __bf16 v16bf;
typedef __attribute__((ext_vector_type(8)))  __bf16 v8bf;
typedef __attribute__((ext_vector_type(8)))  float  v8f;

#define TSEQ  1024
#define NBATCH 2
#define CDIM  768
#define NHEAD 12
#define HDIM  64
#define NROWS (NBATCH * TSEQ)   // 2048
#define VOCAB 50257

static __device__ __forceinline__ v16bf cat8(v8bf lo, v8bf hi) {
  return __builtin_shufflevector(lo, hi, 0,1,2,3,4,5,6,7,8,9,10,11,12,13,14,15);
}

static __device__ __forceinline__ v8f wmma_bf16(v16bf a, v16bf b, v8f c) {
  // (neg_a, A, neg_b, B, c_mod, C, reuse_a, reuse_b)
  return __builtin_amdgcn_wmma_f32_16x16x32_bf16(false, a, false, b, (short)0, c,
                                                 false, false);
}

// async copy 16B global -> LDS (gfx1250, tracked by ASYNCcnt, no VGPR data)
static __device__ __forceinline__ void async_b128(const void* g, void* lds) {
  unsigned l = (unsigned)(uintptr_t)lds;               // low 32 bits = LDS offset
  unsigned long long ga = (unsigned long long)(uintptr_t)g;
  asm volatile("global_load_async_to_lds_b128 %0, %1, off"
               :: "v"(l), "v"(ga) : "memory");
}
static __device__ __forceinline__ void wait_async0() {
  asm volatile("s_wait_asynccnt 0x0" ::: "memory");
}

static __device__ __forceinline__ unsigned pack2(bf16 lo, bf16 hi) {
  unsigned a = __builtin_bit_cast(unsigned short, lo);
  unsigned b = __builtin_bit_cast(unsigned short, hi);
  return a | (b << 16);
}

// ---------------------------------------------------------------------------
// Embedding: x[b,t,:] = tok_emb[idx[b,t],:] + pos_emb[t,:]
// ---------------------------------------------------------------------------
__global__ __launch_bounds__(256)
void gpt_embed(const int* __restrict__ idx, const float* __restrict__ tok,
               const float* __restrict__ pos, float* __restrict__ x)
{
  int i = blockIdx.x * 256 + threadIdx.x;
  if (i >= NROWS * CDIM) return;
  int row = i / CDIM, c = i - row * CDIM;
  int t = row & (TSEQ - 1);
  x[i] = tok[(size_t)idx[row] * CDIM + c] + pos[(size_t)t * CDIM + c];
}

// ---------------------------------------------------------------------------
// LayerNorm (f32 in) -> bf16 out.  One block per row.
// ---------------------------------------------------------------------------
__global__ __launch_bounds__(256)
void gpt_layernorm(const float* __restrict__ x, const float* __restrict__ w,
                   const float* __restrict__ b, bf16* __restrict__ out)
{
  int row = blockIdx.x;
  const float* xr = x + (size_t)row * CDIM;
  int tid = threadIdx.x, lane = tid & 31, wave = tid >> 5;
  __shared__ float sm[8];

  float s = 0.f;
  for (int i = tid; i < CDIM; i += 256) s += xr[i];
  #pragma unroll
  for (int o = 16; o > 0; o >>= 1) s += __shfl_xor(s, o, 32);
  if (lane == 0) sm[wave] = s;
  __syncthreads();
  float tot = 0.f;
  #pragma unroll
  for (int i = 0; i < 8; i++) tot += sm[i];
  float mean = tot * (1.0f / CDIM);
  __syncthreads();

  float v = 0.f;
  for (int i = tid; i < CDIM; i += 256) { float d = xr[i] - mean; v += d * d; }
  #pragma unroll
  for (int o = 16; o > 0; o >>= 1) v += __shfl_xor(v, o, 32);
  if (lane == 0) sm[wave] = v;
  __syncthreads();
  float vt = 0.f;
  #pragma unroll
  for (int i = 0; i < 8; i++) vt += sm[i];
  float rstd = rsqrtf(vt * (1.0f / CDIM) + 1e-5f);

  for (int i = tid; i < CDIM; i += 256)
    out[(size_t)row * CDIM + i] = (bf16)((xr[i] - mean) * rstd * w[i] + b[i]);
}

// ---------------------------------------------------------------------------
// Tiled WMMA GEMM:  C[M,N] = A[M,K](bf16) * B(f32 -> bf16 in LDS)
//   BNMAJOR=false: B stored [K,N] row-major (weights)
//   BNMAJOR=true : B stored [N,K] row-major (tied head: tok_emb), N ragged
// Epilogues: 0 plain f32, 1 bias->bf16, 2 bias+gelu->bf16, 3 bias+resid(f32+=)
// Block tile 128x128x32, 256 threads = 8 waves, wave tile 32x64 (2x4 WMMA).
// Double-buffered LDS; A staged via async global->LDS copies.
// ---------------------------------------------------------------------------
enum { EPI_PLAIN = 0, EPI_BIAS_BF16 = 1, EPI_BIAS_GELU = 2, EPI_RESID = 3 };

template <int EPI, bool BNMAJOR>
__global__ __launch_bounds__(256)
void gpt_gemm(const bf16* __restrict__ A, const float* __restrict__ B,
              const float* __restrict__ bias, void* __restrict__ Cout,
              int M, int N, int K)
{
  constexpr int BM = 128, BN = 128, BK = 32, LS = 40;  // LS: padded LDS stride
  __shared__ bf16 As[2][BM * LS];
  __shared__ bf16 Bs[2][BN * LS];

  const int tid = threadIdx.x, lane = tid & 31, wave = tid >> 5;
  const int half = lane >> 4, l16 = lane & 15;
  const int bm = blockIdx.y * BM, bn = blockIdx.x * BN;
  const int wml = (wave >> 1) * 32;   // wave M offset in tile
  const int wnl = (wave & 1) * 64;    // wave N offset in tile

  v8f acc[2][4] = {};

  auto stageA = [&](int buf, int k0) {
    int r = tid >> 1, cs = (tid & 1) * 16;
    const bf16* g = A + (size_t)(bm + r) * K + k0 + cs;
    bf16* l = &As[buf][r * LS + cs];
    async_b128(g, l);
    async_b128(g + 8, l + 8);
  };

  auto stageB = [&](int buf, int k0) {
    if (BNMAJOR) {
      // B[N][K]: rows load contiguously; convert f32->bf16, vector store
      int nl = tid >> 1, cs = (tid & 1) * 16;
      int n = bn + nl;
      float tmp[16];
      if (n < N) {
        const float4* src = (const float4*)(B + (size_t)n * K + k0 + cs);
        float4* t4 = (float4*)tmp;
        t4[0] = src[0]; t4[1] = src[1]; t4[2] = src[2]; t4[3] = src[3];
      } else {
        #pragma unroll
        for (int j = 0; j < 16; j++) tmp[j] = 0.f;
      }
      v8bf lo, hi;
      #pragma unroll
      for (int j = 0; j < 8; j++) { lo[j] = (bf16)tmp[j]; hi[j] = (bf16)tmp[j + 8]; }
      *(v8bf*)&Bs[buf][nl * LS + cs]     = lo;
      *(v8bf*)&Bs[buf][nl * LS + cs + 8] = hi;
    } else {
      // B[K][N]: load 2 k-rows x 8 n, pack bf16 pairs, store u32 transposed
      int kp = (tid >> 4) * 2, ng = (tid & 15) * 8;
      const float4* r0 = (const float4*)(B + (size_t)(k0 + kp) * N + bn + ng);
      const float4* r1 = (const float4*)(B + (size_t)(k0 + kp + 1) * N + bn + ng);
      float t0[8], t1[8];
      ((float4*)t0)[0] = r0[0]; ((float4*)t0)[1] = r0[1];
      ((float4*)t1)[0] = r1[0]; ((float4*)t1)[1] = r1[1];
      #pragma unroll
      for (int j = 0; j < 8; j++) {
        unsigned u = pack2((bf16)t0[j], (bf16)t1[j]);
        *(unsigned*)&Bs[buf][(ng + j) * LS + kp] = u;
      }
      if (k0 + 2 * BK < K)
        __builtin_prefetch(B + (size_t)(k0 + 2 * BK + kp) * N + bn + ng, 0, 1);
    }
  };

  const int nk = K / BK;
  stageA(0, 0);
  stageB(0, 0);

  for (int ik = 0; ik < nk; ik++) {
    const int buf = ik & 1;
    wait_async0();          // own async copies into `buf` done
    __syncthreads();        // everyone's copies visible, prev compute finished
    if (ik + 1 < nk) { stageA(buf ^ 1, (ik + 1) * BK); stageB(buf ^ 1, (ik + 1) * BK); }

    // ---- A fragments (ISA 16-bit A layout: lanes<16 K{0..7,16..23})
    v16bf af0, af1;
    {
      const bf16* p0 = &As[buf][(wml + l16) * LS + half * 8];
      af0 = cat8(*(const v8bf*)p0, *(const v8bf*)(p0 + 16));
      const bf16* p1 = &As[buf][(wml + 16 + l16) * LS + half * 8];
      af1 = cat8(*(const v8bf*)p1, *(const v8bf*)(p1 + 16));
    }
    #pragma unroll
    for (int ni = 0; ni < 4; ni++) {
      // B fragment: lane = column; lanes<16 hold K{0..7,16..23}, lanes>=16 K{8..15,24..31}
      const bf16* pb = &Bs[buf][(wnl + ni * 16 + l16) * LS + half * 8];
      v16bf bfr = cat8(*(const v8bf*)pb, *(const v8bf*)(pb + 16));
      acc[0][ni] = wmma_bf16(af0, bfr, acc[0][ni]);
      acc[1][ni] = wmma_bf16(af1, bfr, acc[1][ni]);
    }
  }

  // ---- epilogue (C layout: VGPR r -> row r (lanes 0-15) / row r+8 (16-31))
  #pragma unroll
  for (int mi = 0; mi < 2; mi++)
    #pragma unroll
    for (int ni = 0; ni < 4; ni++)
      #pragma unroll
      for (int r = 0; r < 8; r++) {
        int row = bm + wml + mi * 16 + r + half * 8;
        int col = bn + wnl + ni * 16 + l16;
        if (BNMAJOR && col >= N) continue;
        float v = acc[mi][ni][r];
        size_t o = (size_t)row * N + col;
        if (EPI == EPI_PLAIN) {
          ((float*)Cout)[o] = v;
        } else if (EPI == EPI_BIAS_BF16) {
          ((bf16*)Cout)[o] = (bf16)(v + bias[col]);
        } else if (EPI == EPI_BIAS_GELU) {
          float g = v + bias[col];
          ((bf16*)Cout)[o] = (bf16)(0.5f * g * (1.0f + erff(g * 0.70710678f)));
        } else {
          ((float*)Cout)[o] += v + bias[col];
        }
      }
}

// ---------------------------------------------------------------------------
// Flash-style causal attention with WMMA. 128 threads = 4 waves; each wave
// owns a 16-row query tile (WG covers 64 queries of one (b,h)).
// Key blocks of 32 double-buffered through LDS (K via async copies).
// ---------------------------------------------------------------------------
__global__ __launch_bounds__(128)
void gpt_attention(const bf16* __restrict__ qkv, bf16* __restrict__ y)
{
  constexpr int KS = 72;  // K tile LDS stride (bf16 elems)
  constexpr int VS = 40;  // V^T tile LDS stride
  constexpr int PS = 40;  // P tile LDS stride
  __shared__ bf16 Ks[2][32 * KS];     // [key][c]
  __shared__ bf16 Vt[2][64 * VS];     // [c][key]
  __shared__ bf16 Ps[4][16 * PS];     // per-wave [row][key]

  const int tid = threadIdx.x, lane = tid & 31, wave = tid >> 5;
  const int half = lane >> 4, l16 = lane & 15;
  const int b = blockIdx.z, h = blockIdx.y;
  const int qbase = blockIdx.x * 64;
  const int q0 = qbase + wave * 16;

  const size_t rs = 3 * CDIM;  // qkv row stride (elements)
  const bf16* Q  = qkv + (size_t)b * TSEQ * rs + (size_t)h * HDIM;
  const bf16* Kp = Q + CDIM;
  const bf16* Vp = Q + 2 * CDIM;

  // Q fragments: 16x64 = two 16x32 A fragments
  v16bf qa[2];
  #pragma unroll
  for (int f = 0; f < 2; f++) {
    const bf16* qr = Q + (size_t)(q0 + l16) * rs + f * 32 + half * 8;
    qa[f] = cat8(*(const v8bf*)qr, *(const v8bf*)(qr + 16));
  }

  v8f yacc[4] = {};
  float mrow[8], lrow[8];
  #pragma unroll
  for (int r = 0; r < 8; r++) { mrow[r] = -1e30f; lrow[r] = 0.f; }

  auto stageKV = [&](int buf, int kb) {
    // K tile [32][64] bf16: raw async copies
    {
      int kr = tid >> 2, cs = (tid & 3) * 16;
      const bf16* g = Kp + (size_t)(kb + kr) * rs + cs;
      bf16* l = &Ks[buf][kr * KS + cs];
      async_b128(g, l);
      async_b128(g + 8, l + 8);
    }
    // V tile transposed [c][key]: pack key pairs into u32 stores
    {
      int kp = (tid >> 3) * 2, cg = (tid & 7) * 8;
      v8bf r0 = *(const v8bf*)(Vp + (size_t)(kb + kp) * rs + cg);
      v8bf r1 = *(const v8bf*)(Vp + (size_t)(kb + kp + 1) * rs + cg);
      #pragma unroll
      for (int j = 0; j < 8; j++) {
        unsigned u = pack2(r0[j], r1[j]);
        *(unsigned*)&Vt[buf][(cg + j) * VS + kp] = u;
      }
    }
  };

  const int kend = qbase + 64;   // causal bound for this WG
  const int nblk = kend / 32;
  stageKV(0, 0);

  for (int ib = 0; ib < nblk; ib++) {
    const int buf = ib & 1;
    const int kb = ib * 32;
    wait_async0();
    __syncthreads();
    if (ib + 1 < nblk) stageKV(buf ^ 1, kb + 32);

    if (kb <= q0 + 15) {  // wave-uniform: this wave has live keys here
      // ---- S = Q K^T : two 16x16 tiles (keys kb..+15, kb+16..+31)
      v8f s0 = {}, s1 = {};
      #pragma unroll
      for (int f = 0; f < 2; f++) {
        const bf16* pb0 = &Ks[buf][l16 * KS + f * 32 + half * 8];
        v16bf b0 = cat8(*(const v8bf*)pb0, *(const v8bf*)(pb0 + 16));
        s0 = wmma_bf16(qa[f], b0, s0);
        const bf16* pb1 = &Ks[buf][(16 + l16) * KS + f * 32 + half * 8];
        v16bf b1 = cat8(*(const v8bf*)pb1, *(const v8bf*)(pb1 + 16));
        s1 = wmma_bf16(qa[f], b1, s1);
      }
      // ---- online softmax
      const float sc = 0.125f;  // 1/sqrt(64)
      int key0 = kb + l16, key1 = key0 + 16;
      float p0[8], p1[8];
      #pragma unroll
      for (int r = 0; r < 8; r++) {
        int row = q0 + r + half * 8;
        float a  = (key0 > row) ? -1e30f : s0[r] * sc;
        float bb = (key1 > row) ? -1e30f : s1[r] * sc;
        float t = fmaxf(a, bb);
        #pragma unroll
        for (int o = 1; o < 16; o <<= 1) t = fmaxf(t, __shfl_xor(t, o, 32));
        float mn = fmaxf(mrow[r], t);
        float alpha = __expf(mrow[r] - mn);
        mrow[r] = mn;
        lrow[r] *= alpha;
        #pragma unroll
        for (int ni = 0; ni < 4; ni++) yacc[ni][r] *= alpha;
        float e0 = __expf(a - mn), e1 = __expf(bb - mn);
        p0[r] = e0; p1[r] = e1;
        float ssum = e0 + e1;
        #pragma unroll
        for (int o = 1; o < 16; o <<= 1) ssum += __shfl_xor(ssum, o, 32);
        lrow[r] += ssum;
      }
      // ---- P through LDS: C-layout -> A-layout (wave-local, DS in-order)
      bf16* P = &Ps[wave][0];
      #pragma unroll
      for (int r = 0; r < 8; r++) {
        int pr = r + half * 8;
        P[pr * PS + l16]      = (bf16)p0[r];
        P[pr * PS + 16 + l16] = (bf16)p1[r];
      }
      const bf16* pp = &Ps[wave][l16 * PS + half * 8];
      v16bf pa = cat8(*(const v8bf*)pp, *(const v8bf*)(pp + 16));
      // ---- Y += P(16x32) @ V(32x64)
      #pragma unroll
      for (int nc = 0; nc < 4; nc++) {
        const bf16* vb = &Vt[buf][(nc * 16 + l16) * VS + half * 8];
        v16bf bv = cat8(*(const v8bf*)vb, *(const v8bf*)(vb + 16));
        yacc[nc] = wmma_bf16(pa, bv, yacc[nc]);
      }
    }
  }

  // ---- normalize + write y[b, t, h*64 + c] (bf16)
  bf16* out = y + (size_t)b * TSEQ * CDIM + (size_t)h * HDIM;
  #pragma unroll
  for (int nc = 0; nc < 4; nc++)
    #pragma unroll
    for (int r = 0; r < 8; r++) {
      int row = q0 + r + half * 8;
      int col = nc * 16 + l16;
      out[(size_t)row * CDIM + col] = (bf16)(yacc[nc][r] / lrow[r]);
    }
}

// ---------------------------------------------------------------------------
// Host-side orchestration (graph-capture safe: only kernel launches on stream)
// ---------------------------------------------------------------------------
extern "C" void kernel_launch(void* const* d_in, const int* in_sizes, int n_in,
                              void* d_out, int out_size, void* d_ws, size_t ws_size,
                              hipStream_t stream) {
  (void)in_sizes; (void)n_in; (void)out_size; (void)ws_size;
  const int*   idx     = (const int*)  d_in[0];
  const float* tok_emb = (const float*)d_in[1];
  const float* pos_emb = (const float*)d_in[2];
  const float* ln1_w   = (const float*)d_in[3];
  const float* ln1_b   = (const float*)d_in[4];
  const float* w_qkv   = (const float*)d_in[5];
  const float* b_qkv   = (const float*)d_in[6];
  const float* w_proj  = (const float*)d_in[7];
  const float* b_proj  = (const float*)d_in[8];
  const float* ln2_w   = (const float*)d_in[9];
  const float* ln2_b   = (const float*)d_in[10];
  const float* w_fc    = (const float*)d_in[11];
  const float* b_fc    = (const float*)d_in[12];
  const float* w_fc2   = (const float*)d_in[13];
  const float* b_fc2   = (const float*)d_in[14];
  const float* lnf_w   = (const float*)d_in[15];
  const float* lnf_b   = (const float*)d_in[16];
  float* logits = (float*)d_out;

  // workspace carve-up (~35 MB)
  char* w = (char*)d_ws;
  float* xf   = (float*)w; w += (size_t)NROWS * CDIM * 4;        // residual f32
  bf16*  hb   = (bf16*) w; w += (size_t)NROWS * CDIM * 2;        // LN out bf16
  bf16*  qkvb = (bf16*) w; w += (size_t)NROWS * 3 * CDIM * 2;    // qkv bf16
  bf16*  attb = (bf16*) w; w += (size_t)NROWS * CDIM * 2;        // attn out bf16
  bf16*  gelb = (bf16*) w; w += (size_t)NROWS * 4 * CDIM * 2;    // gelu out bf16

  dim3 blk256(256), blk128(128);
  dim3 gEmb((NROWS * CDIM + 255) / 256);
  dim3 gLN(NROWS);
  dim3 gQKV(3 * CDIM / 128, NROWS / 128);
  dim3 gC(CDIM / 128, NROWS / 128);
  dim3 gFC(4 * CDIM / 128, NROWS / 128);
  dim3 gAtt(TSEQ / 64, NHEAD, NBATCH);
  dim3 gHead((VOCAB + 127) / 128, NROWS / 128);

  gpt_embed<<<gEmb, blk256, 0, stream>>>(idx, tok_emb, pos_emb, xf);

  for (int l = 0; l < 12; l++) {
    gpt_layernorm<<<gLN, blk256, 0, stream>>>(xf, ln1_w + (size_t)l * CDIM,
                                              ln1_b + (size_t)l * CDIM, hb);
    gpt_gemm<EPI_BIAS_BF16, false><<<gQKV, blk256, 0, stream>>>(
        hb, w_qkv + (size_t)l * CDIM * 3 * CDIM, b_qkv + (size_t)l * 3 * CDIM,
        qkvb, NROWS, 3 * CDIM, CDIM);
    gpt_attention<<<gAtt, blk128, 0, stream>>>(qkvb, attb);
    gpt_gemm<EPI_RESID, false><<<gC, blk256, 0, stream>>>(
        attb, w_proj + (size_t)l * CDIM * CDIM, b_proj + (size_t)l * CDIM,
        xf, NROWS, CDIM, CDIM);
    gpt_layernorm<<<gLN, blk256, 0, stream>>>(xf, ln2_w + (size_t)l * CDIM,
                                              ln2_b + (size_t)l * CDIM, hb);
    gpt_gemm<EPI_BIAS_GELU, false><<<gFC, blk256, 0, stream>>>(
        hb, w_fc + (size_t)l * CDIM * 4 * CDIM, b_fc + (size_t)l * 4 * CDIM,
        gelb, NROWS, 4 * CDIM, CDIM);
    gpt_gemm<EPI_RESID, false><<<gC, blk256, 0, stream>>>(
        gelb, w_fc2 + (size_t)l * 4 * CDIM * CDIM, b_fc2 + (size_t)l * CDIM,
        xf, NROWS, CDIM, 4 * CDIM);
  }

  gpt_layernorm<<<gLN, blk256, 0, stream>>>(xf, lnf_w, lnf_b, hb);
  // tied head: logits = h @ tok_emb^T ; tok_emb is [V,C] => N-major B
  gpt_gemm<EPI_PLAIN, true><<<gHead, blk256, 0, stream>>>(
      hb, tok_emb, nullptr, logits, NROWS, VOCAB, CDIM);
}
